// CDCK2_1649267441916
// MI455X (gfx1250) — matrix operand order repaired and checked by reading, not compiled, not run on hardware
//
#include <hip/hip_runtime.h>

typedef __attribute__((ext_vector_type(16))) __bf16 v16bf;
typedef __attribute__((ext_vector_type(8)))  __bf16 v8bf;
typedef __attribute__((ext_vector_type(8)))  float  v8f;
typedef __attribute__((ext_vector_type(4)))  unsigned int u32x4;
typedef __attribute__((ext_vector_type(8)))  int i32x8;
typedef __attribute__((ext_vector_type(4)))  int i32x4;

static __device__ __forceinline__ v16bf cat16(v8bf lo, v8bf hi) {
  return __builtin_shufflevector(lo, hi, 0, 1, 2, 3, 4, 5, 6, 7, 8, 9, 10, 11, 12, 13, 14, 15);
}

// exact divmod for 0 <= n < 2^20 using inv = floor(2^32 / d): guess is low by <=1
static __device__ __forceinline__ void divmod_u(int n, int d, unsigned inv, int& q, int& r) {
  q = (int)(((unsigned long long)(unsigned)n * inv) >> 32);
  r = n - q * d;
  if (r >= d) { ++q; r -= d; }
}

// ---------------------------------------------------------------- utilities
__global__ void k_zero(float* __restrict__ p, int n) {
  int i = blockIdx.x * 256 + threadIdx.x;
  if (i < n) p[i] = 0.f;
}

__global__ void k_f32_to_bf16(const float* __restrict__ src, __bf16* __restrict__ dst, int n) {
  int i = blockIdx.x * 256 + threadIdx.x;
  if (i < n) dst[i] = (__bf16)src[i];
}

// ------------------------------------------------- conv1 (ch_in=1) BN stats
__global__ void k_conv1_stats(const float* __restrict__ x, const float* __restrict__ w1,
                              float* __restrict__ sums, int chunk) {
  int c = blockIdx.x;
  float wv[10];
#pragma unroll
  for (int k = 0; k < 10; ++k) wv[k] = w1[c * 10 + k];
  int start = blockIdx.y * chunk;
  float s = 0.f, sq = 0.f;
  for (int p = start + threadIdx.x; p < start + chunk; p += 256) {
    int b = p / 16000, t = p - b * 16000;   // constant divisor -> strength-reduced
    const float* xb = x + b * 80000;
    int base = 5 * t - 3;
    float acc = 0.f;
#pragma unroll
    for (int k = 0; k < 10; ++k) {
      int col = base + k;
      int colc = col < 0 ? 0 : (col > 79999 ? 79999 : col);
      float xv = (col == colc) ? xb[colc] : 0.f;
      acc = fmaf(xv, wv[k], acc);
    }
    s += acc; sq += acc * acc;
  }
  __shared__ float rs[256], rq[256];
  rs[threadIdx.x] = s; rq[threadIdx.x] = sq;
  __syncthreads();
  for (int o = 128; o > 0; o >>= 1) {
    if (threadIdx.x < o) { rs[threadIdx.x] += rs[threadIdx.x + o]; rq[threadIdx.x] += rq[threadIdx.x + o]; }
    __syncthreads();
  }
  if (threadIdx.x == 0) { atomicAdd(&sums[c], rs[0]); atomicAdd(&sums[512 + c], rq[0]); }
}

__global__ void k_bnparam(const float* __restrict__ sums, const float* __restrict__ gamma,
                          const float* __restrict__ beta, int layer, float cnt,
                          float* __restrict__ ss) {
  int c = blockIdx.x * 256 + threadIdx.x;
  if (c >= 512) return;
  float mean = sums[c] / cnt;
  float var  = sums[512 + c] / cnt - mean * mean;
  float sc   = gamma[layer * 512 + c] * rsqrtf(var + 1e-5f);
  ss[c] = sc;
  ss[512 + c] = beta[layer * 512 + c] - mean * sc;
}

// recompute conv1, apply BN+ReLU, emit bf16 (B,512,16000)
__global__ void k_conv1_apply(const float* __restrict__ x, const float* __restrict__ w1,
                              const float* __restrict__ ss, __bf16* __restrict__ out) {
  int o = blockIdx.x * 256 + threadIdx.x;
  int t = o % 16000;
  int rem = o / 16000;
  int c = rem & 511, b = rem >> 9;
  const float* xb = x + b * 80000;
  const float* wc = w1 + c * 10;
  int base = 5 * t - 3;
  float acc = 0.f;
#pragma unroll
  for (int k = 0; k < 10; ++k) {
    int col = base + k;
    int colc = col < 0 ? 0 : (col > 79999 ? 79999 : col);
    float xv = (col == colc) ? xb[colc] : 0.f;
    acc = fmaf(xv, wc[k], acc);
  }
  float v = fmaf(acc, ss[c], ss[512 + c]);
  out[o] = (__bf16)fmaxf(v, 0.f);
}

// ---------------------------------------------- generic raw-f32 BN stats/apply
__global__ void k_stats_raw(const float* __restrict__ raw, float* __restrict__ sums,
                            int L, unsigned invL, int chunk) {
  int c = blockIdx.x;
  int total = 16 * L;
  int start = blockIdx.y * chunk;
  int end = start + chunk; if (end > total) end = total;
  float s = 0.f, sq = 0.f;
  for (int p = start + threadIdx.x; p < end; p += 256) {
    int b, t;
    divmod_u(p, L, invL, b, t);
    float v = raw[((b << 9) + c) * L + t];
    s += v; sq += v * v;
  }
  __shared__ float rs[256], rq[256];
  rs[threadIdx.x] = s; rq[threadIdx.x] = sq;
  __syncthreads();
  for (int o = 128; o > 0; o >>= 1) {
    if (threadIdx.x < o) { rs[threadIdx.x] += rs[threadIdx.x + o]; rq[threadIdx.x] += rq[threadIdx.x + o]; }
    __syncthreads();
  }
  if (threadIdx.x == 0) { atomicAdd(&sums[c], rs[0]); atomicAdd(&sums[512 + c], rq[0]); }
}

__global__ void k_bn_apply(const float* __restrict__ raw, const float* __restrict__ ss,
                           __bf16* __restrict__ out, int L, unsigned invL, int n) {
  int o = blockIdx.x * 256 + threadIdx.x;
  if (o >= n) return;
  int rem, t;
  divmod_u(o, L, invL, rem, t);
  int c = rem & 511;
  float v = fmaf(raw[o], ss[c], ss[512 + c]);
  out[o] = (__bf16)fmaxf(v, 0.f);
}

// variant writing transposed (B, L, 512) layout — used for layer 5 so the GRU
// and head read channel-contiguous rows.
__global__ void k_bn_apply_t(const float* __restrict__ raw, const float* __restrict__ ss,
                             __bf16* __restrict__ out, int L, unsigned invL, int n) {
  int o = blockIdx.x * 256 + threadIdx.x;
  if (o >= n) return;
  int rem, t;
  divmod_u(o, L, invL, rem, t);
  int c = rem & 511, b = rem >> 9;
  float v = fmaf(raw[o], ss[c], ss[512 + c]);
  out[(b * L + t) * 512 + c] = (__bf16)fmaxf(v, 0.f);
}

// --------------------------------------------------- conv2..5 as im2col GEMM
// M=512 oc, N=16*Lout positions (n=b*Lout+t), Kdim=512*ks.
// Block 256 thr = 8 waves; tile 128(M) x 64(N); each wave: 2 M-tiles x 2 N-tiles
// = 4 accumulators, 4 WMMA per K-step.
// A tile (weights, 128x32 bf16) is fetched by the Tensor Data Mover:
// wave 0 issues tensor_load_to_lds with LDS row padding 16DW data + 4DW pad
// (row stride 40 elements), then s_wait_tensorcnt 0 before the barrier.
__global__ void __launch_bounds__(256) k_conv_gemm(
    const __bf16* __restrict__ xin, const __bf16* __restrict__ w,
    float* __restrict__ yraw, int Lin, int Lout, unsigned invLout,
    int ksLog2, int stride, int pad) {
  __shared__ __bf16 sA[128 * 40];
  __shared__ __bf16 sB[64 * 40];
  const int Kdim = 512 << ksLog2;
  const int ks = 1 << ksLog2;
  const int tid = threadIdx.x;
  const int oc0 = blockIdx.y * 128;
  const int n0g = blockIdx.x * 64;
  const int wave = tid >> 5, lane = tid & 31;
  const int hi = (lane >> 4) & 1, lm = lane & 15;
  const int mTile0 = (wave & 3) * 2;        // 2 M-tiles per wave
  const int nTile0 = (wave >> 2) * 2;       // 2 N-tiles per wave
  // per-thread staging coordinates for the B tile
  const int bn = tid >> 2, bk = (tid & 3) * 8;   // B row bn
  int bB, tB;
  divmod_u(n0g + bn, Lout, invLout, bB, tB);

  // TDM descriptor pieces that do not change across K-steps
  const unsigned ldsAddr = (unsigned)(unsigned long long)(void*)sA;
  i32x8 g1 = {};
  g1[0] = (1 << 16) | (1 << 20) | (3 << 22) | (3 << 25); // data_size=2B, pad_en, 16DW interval, 4DW pad
  g1[1] = (int)((unsigned)(Kdim & 0xFFFF) << 16);        // tensor_dim0 lo16
  g1[2] = (int)(((unsigned)Kdim >> 16) | (512u << 16));  // tensor_dim0 hi16 | tensor_dim1 lo16
  g1[3] = (int)(32u << 16);                              // tensor_dim1 hi16=0 | tile_dim0=32
  g1[4] = 128;                                           // tile_dim1=128, tile_dim2=0
  g1[5] = Kdim;                                          // tensor_dim0_stride lo32
  g1[6] = 0;
  g1[7] = 0;
  i32x4 gz = {};

  v8f acc00 = {}; v8f acc01 = {}; v8f acc10 = {}; v8f acc11 = {};
  for (int k0 = 0; k0 < Kdim; k0 += 32) {
    if (wave == 0) {  // A tile via Tensor Data Mover (one wave issues the DMA)
      unsigned long long ga =
          (unsigned long long)(const void*)(w + (size_t)oc0 * Kdim + k0);
      u32x4 g0;
      g0[0] = 1u;                                         // count=1 (valid), user mode
      g0[1] = ldsAddr;                                    // lds_addr (bytes)
      g0[2] = (unsigned)(ga & 0xFFFFFFFFull);             // global_addr lo
      g0[3] = (unsigned)((ga >> 32) & 0x01FFFFFFull) | (2u << 30);  // global_addr hi | type=2
#if defined(__clang_major__) && (__clang_major__ >= 23)
      i32x8 gz8 = {};
      __builtin_amdgcn_tensor_load_to_lds(g0, g1, gz, gz, gz8, 0);
#else
      __builtin_amdgcn_tensor_load_to_lds(g0, g1, gz, gz, 0);
#endif
    }
    {  // B tile 64x32: im2col gather, branch-free (clamped load + select)
      __bf16 vals[8];
#pragma unroll
      for (int i = 0; i < 8; ++i) {
        int kap = k0 + bk + i;
        int ic = kap >> ksLog2, kk = kap & (ks - 1);
        int col = tB * stride - pad + kk;
        int colc = col < 0 ? 0 : (col >= Lin ? Lin - 1 : col);
        __bf16 vv = xin[((bB << 9) + ic) * Lin + colc];
        vals[i] = (col == colc) ? vv : (__bf16)0.0f;
      }
      v8bf v;
#pragma unroll
      for (int i = 0; i < 8; ++i) v[i] = vals[i];
      *(v8bf*)(sB + bn * 40 + bk) = v;
    }
    if (wave == 0) __builtin_amdgcn_s_wait_tensorcnt(0);  // A tile landed in LDS
    __syncthreads();
    const __bf16* ar0 = sA + (mTile0 * 16 + lm) * 40;
    const __bf16* ar1 = ar0 + 16 * 40;
    v16bf a0 = cat16(*(const v8bf*)(ar0 + 16 * hi), *(const v8bf*)(ar0 + 16 * hi + 8));
    v16bf a1 = cat16(*(const v8bf*)(ar1 + 16 * hi), *(const v8bf*)(ar1 + 16 * hi + 8));
    const __bf16* br0 = sB + (nTile0 * 16 + lm) * 40 + hi * 16;
    const __bf16* br1 = br0 + 16 * 40;
    v16bf b0 = cat16(*(const v8bf*)br0, *(const v8bf*)(br0 + 8));
    v16bf b1 = cat16(*(const v8bf*)br1, *(const v8bf*)(br1 + 8));
    acc00 = __builtin_amdgcn_wmma_f32_16x16x32_bf16(false, a0, false, b0, (short)0, acc00, false, false);
    acc01 = __builtin_amdgcn_wmma_f32_16x16x32_bf16(false, a0, false, b1, (short)0, acc01, false, false);
    acc10 = __builtin_amdgcn_wmma_f32_16x16x32_bf16(false, a1, false, b0, (short)0, acc10, false, false);
    acc11 = __builtin_amdgcn_wmma_f32_16x16x32_bf16(false, a1, false, b1, (short)0, acc11, false, false);
    __syncthreads();
  }
  int b0v, t0v, b1v, t1v;
  divmod_u(n0g + nTile0 * 16 + lm, Lout, invLout, b0v, t0v);
  divmod_u(n0g + (nTile0 + 1) * 16 + lm, Lout, invLout, b1v, t1v);
#pragma unroll
  for (int r = 0; r < 8; ++r) {
    int m0 = oc0 + mTile0 * 16 + r + 8 * hi;
    int m1 = m0 + 16;
    yraw[((b0v << 9) + m0) * Lout + t0v] = acc00[r];
    yraw[((b1v << 9) + m0) * Lout + t1v] = acc01[r];
    yraw[((b0v << 9) + m1) * Lout + t0v] = acc10[r];
    yraw[((b1v << 9) + m1) * Lout + t1v] = acc11[r];
  }
}

// ----------------------------------------------------------------- GRU scan
// x5t is (B, 500, 512) bf16 (channel-contiguous). One workgroup, 8 waves.
__global__ void __launch_bounds__(256) k_gru(
    const __bf16* __restrict__ x5t, const __bf16* __restrict__ wih,
    const __bf16* __restrict__ whh, const float* __restrict__ bih,
    const float* __restrict__ bhh, const float* __restrict__ h0,
    const int* __restrict__ tsamp, float* __restrict__ ct, float* __restrict__ dout) {
  extern __shared__ unsigned char smem[];
  float*  h_f = (float*)smem;                                  // 16*256 f32
  __bf16* h_b = (__bf16*)(smem + 16384);                       // 16*256 bf16
  __bf16* xt  = (__bf16*)(smem + 16384 + 8192);                // 16*512 bf16
  float*  Gi  = (float*)(smem + 16384 + 8192 + 16384);         // 16*768 f32
  float*  Gh  = Gi + 16 * 768;                                 // 16*768 f32
  __shared__ int ts_s[16];
  const int tid = threadIdx.x;
  const int wave = tid >> 5, lane = tid & 31;
  const int hi = (lane >> 4) & 1, lm = lane & 15;
  if (tid < 16) ts_s[tid] = tsamp[tid];
  for (int i = tid; i < 4096; i += 256) h_f[i] = h0[i];
  __syncthreads();
  const int xb = tid >> 4;               // batch row this thread stages
  const int xc0 = (tid & 15) * 32;       // 32 contiguous channels
  const int hb0 = tid * 16;              // h_b conversion chunk
  for (int t = 0; t < 500; ++t) {
    {  // stage masked x_t: 4x v8bf per thread
      v8bf z = {};
      bool act = (t <= ts_s[xb]);
      const __bf16* src = x5t + (xb * 500 + t) * 512 + xc0;
      __builtin_prefetch(src + 512, 0, 3);           // next timestep row
#pragma unroll
      for (int i = 0; i < 4; ++i) {
        v8bf v = z;
        if (act) v = *(const v8bf*)(src + i * 8);
        *(v8bf*)(xt + xb * 512 + xc0 + i * 8) = v;
      }
    }
#pragma unroll
    for (int i = 0; i < 16; ++i) h_b[hb0 + i] = (__bf16)h_f[hb0 + i];
    __syncthreads();
    for (int nt = 0; nt < 6; ++nt) {
      int n0 = (wave * 6 + nt) * 16;
      v8f acc = {};
      for (int k0 = 0; k0 < 512; k0 += 32) {
        const __bf16* ar = xt + lm * 512 + k0;
        v16bf a = cat16(*(const v8bf*)(ar + 16 * hi), *(const v8bf*)(ar + 16 * hi + 8));
        const __bf16* br = wih + (n0 + lm) * 512 + k0 + hi * 16;
        v16bf bm = cat16(*(const v8bf*)br, *(const v8bf*)(br + 8));
        acc = __builtin_amdgcn_wmma_f32_16x16x32_bf16(false, a, false, bm, (short)0, acc, false, false);
      }
#pragma unroll
      for (int r = 0; r < 8; ++r) Gi[(r + 8 * hi) * 768 + n0 + lm] = acc[r];
      v8f acch = {};
      for (int k0 = 0; k0 < 256; k0 += 32) {
        const __bf16* ar = h_b + lm * 256 + k0;
        v16bf a = cat16(*(const v8bf*)(ar + 16 * hi), *(const v8bf*)(ar + 16 * hi + 8));
        const __bf16* br = whh + (n0 + lm) * 256 + k0 + hi * 16;
        v16bf bm = cat16(*(const v8bf*)br, *(const v8bf*)(br + 8));
        acch = __builtin_amdgcn_wmma_f32_16x16x32_bf16(false, a, false, bm, (short)0, acch, false, false);
      }
#pragma unroll
      for (int r = 0; r < 8; ++r) Gh[(r + 8 * hi) * 768 + n0 + lm] = acch[r];
    }
    __syncthreads();
    for (int i = tid; i < 4096; i += 256) {                    // gate math
      int b = i >> 8, j = i & 255;
      float ir = Gi[b * 768 + j]       + bih[j];
      float iz = Gi[b * 768 + 256 + j] + bih[256 + j];
      float in_= Gi[b * 768 + 512 + j] + bih[512 + j];
      float hr = Gh[b * 768 + j]       + bhh[j];
      float hz = Gh[b * 768 + 256 + j] + bhh[256 + j];
      float hn = Gh[b * 768 + 512 + j] + bhh[512 + j];
      float r  = 1.f / (1.f + __expf(-(ir + hr)));
      float zg = 1.f / (1.f + __expf(-(iz + hz)));
      float nn = tanhf(in_ + r * hn);
      float hnew = (1.f - zg) * nn + zg * h_f[i];
      h_f[i] = hnew;
      if (t == ts_s[b]) ct[i] = hnew;
    }
    __syncthreads();
  }
  for (int i = tid; i < 4096; i += 256) dout[2 + i] = h_f[i];
}

// ----------------------------------------------------- contrastive head
__global__ void __launch_bounds__(256) k_final(
    const float* __restrict__ ct, const __bf16* __restrict__ x5t,
    const float* __restrict__ WkW, const float* __restrict__ WkB,
    const int* __restrict__ tsamp, float* __restrict__ dout) {
  __shared__ float pred[16 * 512];
  __shared__ float tot[256];
  __shared__ float nce_b[16];
  __shared__ float rowmax[16], rowsum[16];
  __shared__ int correct;
  const int tid = threadIdx.x;
  if (tid < 16) nce_b[tid] = 0.f;
  if (tid == 0) correct = 0;
  __syncthreads();
  for (int k = 0; k < 12; ++k) {
    for (int idx = tid; idx < 8192; idx += 256) {        // pred[k,c,d]
      int c = idx >> 9, d = idx & 511;
      float s = WkB[k * 512 + d];
      const float* wr = WkW + ((k * 512) + d) * 256;
      const float* cr = ct + c * 256;
      for (int h = 0; h < 256; ++h) s = fmaf(cr[h], wr[h], s);
      pred[idx] = s;
    }
    __syncthreads();
    {                                                     // total[k,b,c]
      int b = tid >> 4, c = tid & 15;
      int te = tsamp[b] + k + 1;
      float s = 0.f;
      const float* pr = pred + c * 512;
      const __bf16* er = x5t + (b * 500 + te) * 512;
      for (int d = 0; d < 512; ++d) s = fmaf((float)er[d], pr[d], s);
      tot[tid] = s;
    }
    __syncthreads();
    if (tid < 16) {                                       // log-softmax row b
      int b = tid;
      float m = tot[b * 16];
      for (int c = 1; c < 16; ++c) m = fmaxf(m, tot[b * 16 + c]);
      float se = 0.f;
      for (int c = 0; c < 16; ++c) se += __expf(tot[b * 16 + c] - m);
      nce_b[b] += tot[b * 16 + b] - (m + __logf(se));
      if (k == 11) { rowmax[b] = m; rowsum[b] = se; }
    }
    __syncthreads();
    if (k == 11 && tid < 16) {                            // accuracy
      int c = tid;
      float best = -1.f; int bi = 0;
      for (int b = 0; b < 16; ++b) {
        float v = __expf(tot[b * 16 + c] - rowmax[b]) / rowsum[b];
        if (v > best) { best = v; bi = b; }
      }
      if (bi == c) atomicAdd(&correct, 1);
    }
    __syncthreads();
  }
  if (tid == 0) {
    float nce = 0.f;
    for (int b = 0; b < 16; ++b) nce += nce_b[b];
    dout[1] = nce / (-1.f * 16.f * 12.f);
    dout[0] = (float)correct / 16.f;
  }
}

// ------------------------------------------------------------------ launch
extern "C" void kernel_launch(void* const* d_in, const int* in_sizes, int n_in,
                              void* d_out, int out_size, void* d_ws, size_t ws_size,
                              hipStream_t stream) {
  (void)in_sizes; (void)n_in; (void)out_size; (void)ws_size;
  const float* x      = (const float*)d_in[0];
  const float* hidden = (const float*)d_in[1];
  const int*   tsamp  = (const int*)d_in[3];
  const float* w1     = (const float*)d_in[4];
  const float* w2     = (const float*)d_in[5];
  const float* w345   = (const float*)d_in[6];
  const float* gamma  = (const float*)d_in[7];
  const float* beta   = (const float*)d_in[8];
  const float* wih    = (const float*)d_in[9];
  const float* whh    = (const float*)d_in[10];
  const float* bih    = (const float*)d_in[11];
  const float* bhh    = (const float*)d_in[12];
  const float* WkW    = (const float*)d_in[13];
  const float* WkB    = (const float*)d_in[14];
  float* dout = (float*)d_out;

  char* ws = (char*)d_ws;
  size_t off = 0;
  auto alloc = [&](size_t bytes) -> char* {
    char* p = ws + off;
    off += (bytes + 255) & ~(size_t)255;
    return p;
  };
  __bf16* w2b  = (__bf16*)alloc((size_t)512 * 4096 * 2);
  __bf16* w3b  = (__bf16*)alloc((size_t)512 * 2048 * 2);
  __bf16* w4b  = (__bf16*)alloc((size_t)512 * 2048 * 2);
  __bf16* w5b  = (__bf16*)alloc((size_t)512 * 2048 * 2);
  __bf16* wihb = (__bf16*)alloc((size_t)768 * 512 * 2);
  __bf16* whhb = (__bf16*)alloc((size_t)768 * 256 * 2);
  __bf16* x1   = (__bf16*)alloc((size_t)16 * 512 * 16000 * 2);
  __bf16* x2   = (__bf16*)alloc((size_t)16 * 512 * 4000 * 2);
  __bf16* x3   = (__bf16*)alloc((size_t)16 * 512 * 2000 * 2);
  __bf16* x4   = (__bf16*)alloc((size_t)16 * 512 * 1000 * 2);
  __bf16* x5t  = (__bf16*)alloc((size_t)16 * 500 * 512 * 2);   // transposed (B,L,C)
  float*  raw  = (float*)alloc((size_t)16 * 512 * 4000 * 4);
  float*  sums = (float*)alloc(1024 * 4);
  float*  ss   = (float*)alloc(1024 * 4);
  float*  ct   = (float*)alloc(16 * 256 * 4);

  // weight conversions f32 -> bf16
  k_f32_to_bf16<<<(512 * 4096 + 255) / 256, 256, 0, stream>>>(w2, w2b, 512 * 4096);
  k_f32_to_bf16<<<(512 * 2048 + 255) / 256, 256, 0, stream>>>(w345 + 0 * 512 * 2048, w3b, 512 * 2048);
  k_f32_to_bf16<<<(512 * 2048 + 255) / 256, 256, 0, stream>>>(w345 + 1 * 512 * 2048, w4b, 512 * 2048);
  k_f32_to_bf16<<<(512 * 2048 + 255) / 256, 256, 0, stream>>>(w345 + 2 * 512 * 2048, w5b, 512 * 2048);
  k_f32_to_bf16<<<(768 * 512 + 255) / 256, 256, 0, stream>>>(wih, wihb, 768 * 512);
  k_f32_to_bf16<<<(768 * 256 + 255) / 256, 256, 0, stream>>>(whh, whhb, 768 * 256);

  // layer 1: stats (recompute), bn params, apply (recompute)
  k_zero<<<4, 256, 0, stream>>>(sums, 1024);
  k_conv1_stats<<<dim3(512, 32), 256, 0, stream>>>(x, w1, sums, 8000);
  k_bnparam<<<2, 256, 0, stream>>>(sums, gamma, beta, 0, 256000.f, ss);
  k_conv1_apply<<<(16 * 512 * 16000) / 256, 256, 0, stream>>>(x, w1, ss, x1);

  // layers 2..5: WMMA GEMM (TDM-fed A tiles) -> stats -> apply
  struct Layer {
    const __bf16* xin; __bf16* xout; const __bf16* w;
    int Lin, Lout, ksLog2, stride, pad, layer; bool transposedOut;
  } Ls[4] = {
    { x1, x2,  w2b, 16000, 4000, 3, 4, 2, 1, false },
    { x2, x3,  w3b,  4000, 2000, 2, 2, 1, 2, false },
    { x3, x4,  w4b,  2000, 1000, 2, 2, 1, 3, false },
    { x4, x5t, w5b,  1000,  500, 2, 2, 1, 4, true  },
  };
  for (int l = 0; l < 4; ++l) {
    const Layer& L = Ls[l];
    unsigned invL = (unsigned)(((unsigned long long)1 << 32) / (unsigned)L.Lout);
    int nTiles = (16 * L.Lout) / 64;
    k_conv_gemm<<<dim3(nTiles, 4), 256, 0, stream>>>(L.xin, L.w, raw, L.Lin, L.Lout, invL,
                                                     L.ksLog2, L.stride, L.pad);
    k_zero<<<4, 256, 0, stream>>>(sums, 1024);
    int total = 16 * L.Lout;
    int chunk = (total + 31) / 32;
    k_stats_raw<<<dim3(512, 32), 256, 0, stream>>>(raw, sums, L.Lout, invL, chunk);
    k_bnparam<<<2, 256, 0, stream>>>(sums, gamma, beta, L.layer, (float)total, ss);
    int n = 16 * 512 * L.Lout;
    if (L.transposedOut)
      k_bn_apply_t<<<(n + 255) / 256, 256, 0, stream>>>(raw, ss, L.xout, L.Lout, invL, n);
    else
      k_bn_apply<<<(n + 255) / 256, 256, 0, stream>>>(raw, ss, L.xout, L.Lout, invL, n);
  }

  // GRU scan (single persistent workgroup, WMMA per step)
  size_t smem = 16384 + 8192 + 16384 + 49152 + 49152;   // 139,264 B < 320 KB/WGP
  k_gru<<<1, 256, smem, stream>>>(x5t, wihb, whhb, bih, bhh, hidden, tsamp, ct, dout);

  // contrastive head: pred/total/nce/accuracy
  k_final<<<1, 256, 0, stream>>>(ct, x5t, WkW, WkB, tsamp, dout);
}
